// NearestEmbedding_22479858827949
// MI455X (gfx1250) — matrix operand order repaired and checked by reading, not compiled, not run on hardware
//
#include <hip/hip_runtime.h>
#include <hip/hip_bf16.h>
#include <stdint.h>

// ---------------------------------------------------------------------------
// NearestEmbedding (VQ) for MI455X / gfx1250:
//   1) batch-norm stats over columns (two-pass, deterministic)
//   2) split codebook w into bf16 hi/lo pairs + ||w_k||^2
//   3) block-cooperative argmin: codebook tiles staged to LDS with
//      GLOBAL_LOAD_ASYNC_TO_LDS_B128 (double-buffered, ASYNCcnt), scores via
//      v_wmma_f32_16x16x32_bf16 with 3-product hi/lo compensation:
//        dist_k = ||w_k||^2 - 2 * xn . w_k   (argmin-equivalent)
//   4) per-row argmin via in-register tracking + wave32 shfl_xor butterfly
//   5) gather codebook rows to output
// ---------------------------------------------------------------------------

typedef __attribute__((ext_vector_type(16))) __bf16          v16bf;
typedef __attribute__((ext_vector_type(16))) unsigned short  v16u;
typedef __attribute__((ext_vector_type(8)))  float           v8f;
typedef __attribute__((ext_vector_type(4)))  int             v4i;

// Pointer-to-AS1(int4) and pointer-to-AS3(int4), as the async builtin expects.
typedef __attribute__((address_space(1))) v4i* gv4i_p;
typedef __attribute__((address_space(3))) v4i* sv4i_p;

#define DIMS 256
#define KT   64          // codes per LDS tile
#define BN_EPS 1e-5f

#if defined(__HIP_DEVICE_COMPILE__) &&                                   \
    __has_builtin(__builtin_amdgcn_global_load_async_to_lds_b128) &&     \
    __has_builtin(__builtin_amdgcn_s_wait_asynccnt)
#define USE_ASYNC 1
#else
#define USE_ASYNC 0
#endif

__device__ __forceinline__ unsigned short f32_to_bf16_bits(float x) {
  unsigned u = __float_as_uint(x);
  unsigned r = u + 0x7FFFu + ((u >> 16) & 1u);   // round-to-nearest-even
  return (unsigned short)(r >> 16);
}
__device__ __forceinline__ float bf16_bits_to_f32(unsigned short b) {
  return __uint_as_float(((unsigned)b) << 16);
}

// ---- Kernel 1: per-column partial sums over 64-row slabs (coalesced) -------
__global__ void k_colstats_partial(const float* __restrict__ x,
                                   float* __restrict__ psum,
                                   float* __restrict__ psq,
                                   int rows_per_block) {
  int d  = threadIdx.x;
  int r0 = blockIdx.x * rows_per_block;
  float s = 0.f, q = 0.f;
  for (int i = 0; i < rows_per_block; ++i) {
    float v = x[(size_t)(r0 + i) * DIMS + d];
    s += v;
    q += v * v;
  }
  psum[(size_t)blockIdx.x * DIMS + d] = s;
  psq [(size_t)blockIdx.x * DIMS + d] = q;
}

// ---- Kernel 2: finalize mean / rstd per column -----------------------------
__global__ void k_colstats_final(const float* __restrict__ psum,
                                 const float* __restrict__ psq,
                                 int nblocks, float invN,
                                 float* __restrict__ meanOut,
                                 float* __restrict__ rstdOut) {
  int d = threadIdx.x;
  float s = 0.f, q = 0.f;
  for (int b = 0; b < nblocks; ++b) {
    s += psum[(size_t)b * DIMS + d];
    q += psq [(size_t)b * DIMS + d];
  }
  float mean = s * invN;
  float var  = q * invN - mean * mean;
  meanOut[d] = mean;
  rstdOut[d] = rsqrtf(var + BN_EPS);
}

// ---- Kernel 3: split codebook into bf16 hi/lo + row squared norms ----------
__global__ void k_split_w(const float* __restrict__ w,
                          unsigned short* __restrict__ wh,
                          unsigned short* __restrict__ wl,
                          float* __restrict__ wsq) {
  __shared__ float red[DIMS];
  int k = blockIdx.x;
  int d = threadIdx.x;
  float v = w[(size_t)k * DIMS + d];
  unsigned short hb = f32_to_bf16_bits(v);
  float lo = v - bf16_bits_to_f32(hb);
  wh[(size_t)k * DIMS + d] = hb;
  wl[(size_t)k * DIMS + d] = f32_to_bf16_bits(lo);
  red[d] = v * v;
  __syncthreads();
  for (int s2 = DIMS / 2; s2 > 0; s2 >>= 1) {
    if (d < s2) red[d] += red[d + s2];
    __syncthreads();
  }
  if (d == 0) wsq[k] = red[0];
}

// ---- Tile staging: global -> LDS, 32KB per array (hi, lo) ------------------
// Each thread moves 8 x 16B chunks per array. Async path uses the CDNA5
// ASYNC-to-LDS DMA; fallback is a plain reg-staged copy.
__device__ __forceinline__ void stage_tile(const unsigned short* __restrict__ wh,
                                           const unsigned short* __restrict__ wl,
                                           unsigned short* dh,
                                           unsigned short* dl,
                                           int k0, int tid) {
  const uint4* sh = (const uint4*)(wh + (size_t)k0 * DIMS);
  const uint4* sl = (const uint4*)(wl + (size_t)k0 * DIMS);
  uint4* th = (uint4*)dh;
  uint4* tl = (uint4*)dl;
  const int chunks = (KT * DIMS) / 8;   // uint4 chunks per array (2048)
#if USE_ASYNC
  #pragma unroll
  for (int i = 0; i < chunks / 256; ++i) {
    int c = i * 256 + tid;
    __builtin_amdgcn_global_load_async_to_lds_b128(
        (gv4i_p)(sh + c), (sv4i_p)(th + c), 0, 0);
    __builtin_amdgcn_global_load_async_to_lds_b128(
        (gv4i_p)(sl + c), (sv4i_p)(tl + c), 0, 0);
  }
#else
  #pragma unroll
  for (int i = 0; i < chunks / 256; ++i) {
    int c = i * 256 + tid;
    th[c] = sh[c];
    tl[c] = sl[c];
  }
#endif
}

__device__ __forceinline__ void stage_wait() {
#if USE_ASYNC
  __builtin_amdgcn_s_wait_asynccnt(0);
#endif
  __syncthreads();
}

// ---- Kernel 4: WMMA argmin -------------------------------------------------
// Block = 256 threads = 8 wave32s; wave w owns rows [blk*128 + w*16, +16).
// Block-level K loop in tiles of KT codes staged in (double-buffered) LDS;
// each wave computes 16x16 score tiles: 8 slices of v_wmma_f32_16x16x32_bf16
// x 3 products (hh, hl, lh) for ~fp32 accuracy over the D=256 contraction.
__global__ void __launch_bounds__(256) k_argmin(
    const float* __restrict__ x,
    const unsigned short* __restrict__ wh,
    const unsigned short* __restrict__ wl,
    const float* __restrict__ wsq,
    const float* __restrict__ meanv,
    const float* __restrict__ rstdv,
    const float* __restrict__ gamma,
    const float* __restrict__ beta,
    int K,
    int* __restrict__ bestIdx) {
  extern __shared__ unsigned short lds[];   // [2 bufs][hi|lo][KT*DIMS]

  int tid  = threadIdx.x;
  int wave = tid >> 5;
  int lane = tid & 31;
  int half = lane >> 4;   // 0: lanes 0-15, 1: lanes 16-31
  int sub  = lane & 15;
  int row  = blockIdx.x * 128 + wave * 16 + sub;

  // Build A operands (16x32 bf16 per slice) for this wave's 16 rows.
  // Lane layout: lanes 0-15 carry K=0..15; lanes 16-31 carry
  // K=16..31 of each 32-wide slice, for row M = lane&15.
  v16bf ah[8], al[8];
  const float* xr = x + (size_t)row * DIMS;
  #pragma unroll
  for (int s = 0; s < 8; ++s) {
    v16u hbits, lbits;
    int base = s * 32 + half * 16;
    #pragma unroll
    for (int e = 0; e < 8; ++e) {
      int d1 = base + e;
      int d2 = d1 + 8;
      float v1 = (xr[d1] - meanv[d1]) * rstdv[d1] * gamma[d1] + beta[d1];
      float v2 = (xr[d2] - meanv[d2]) * rstdv[d2] * gamma[d2] + beta[d2];
      unsigned short h1 = f32_to_bf16_bits(v1);
      unsigned short h2 = f32_to_bf16_bits(v2);
      hbits[e]     = h1;
      hbits[e + 8] = h2;
      lbits[e]     = f32_to_bf16_bits(v1 - bf16_bits_to_f32(h1));
      lbits[e + 8] = f32_to_bf16_bits(v2 - bf16_bits_to_f32(h2));
    }
    ah[s] = __builtin_bit_cast(v16bf, hbits);
    al[s] = __builtin_bit_cast(v16bf, lbits);
  }

  float minv[8];
  int   mini[8];
  #pragma unroll
  for (int j = 0; j < 8; ++j) { minv[j] = 3.4e38f; mini[j] = 0; }

  // B layout (same k order as A): column N = lane&15, lanes 0-15 carry K=0..15, lanes
  // 16-31 carry K=16..31 of the slice -> 16 contiguous bf16 (32 bytes).
  int bcolOff = half * 16;
  int ntiles  = K / KT;

  // Prologue: stage tile 0 into buffer 0.
  stage_tile(wh, wl, lds, lds + KT * DIMS, 0, tid);
  stage_wait();

  for (int t = 0; t < ntiles; ++t) {
    int cur = t & 1;
    unsigned short* curh = lds + (size_t)cur * (2 * KT * DIMS);
    unsigned short* curl = curh + KT * DIMS;

    // Prefetch next tile into the other buffer (overlaps with compute).
    if (t + 1 < ntiles) {
      unsigned short* nxth = lds + (size_t)(cur ^ 1) * (2 * KT * DIMS);
      stage_tile(wh, wl, nxth, nxth + KT * DIMS, (t + 1) * KT, tid);
    }

    #pragma unroll
    for (int kk = 0; kk < KT; kk += 16) {
      int kcol = t * KT + kk + sub;
      float wq = wsq[kcol];
      const unsigned short* brh = curh + (size_t)(kk + sub) * DIMS + bcolOff;
      const unsigned short* brl = curl + (size_t)(kk + sub) * DIMS + bcolOff;

      v8f acc0 = {0.f, 0.f, 0.f, 0.f, 0.f, 0.f, 0.f, 0.f};
      v8f acc1 = {0.f, 0.f, 0.f, 0.f, 0.f, 0.f, 0.f, 0.f};
      v8f acc2 = {0.f, 0.f, 0.f, 0.f, 0.f, 0.f, 0.f, 0.f};
      #pragma unroll
      for (int s = 0; s < 8; ++s) {
        v16bf bh = __builtin_bit_cast(v16bf, *(const v16u*)(brh + s * 32));
        v16bf bl = __builtin_bit_cast(v16bf, *(const v16u*)(brl + s * 32));
        acc0 = __builtin_amdgcn_wmma_f32_16x16x32_bf16(false, ah[s], false, bh,
                                                       (short)0, acc0, false, false);
        acc1 = __builtin_amdgcn_wmma_f32_16x16x32_bf16(false, ah[s], false, bl,
                                                       (short)0, acc1, false, false);
        acc2 = __builtin_amdgcn_wmma_f32_16x16x32_bf16(false, al[s], false, bh,
                                                       (short)0, acc2, false, false);
      }

      // C/D layout: slot j -> row (half ? j+8 : j) of the tile, col = sub.
      #pragma unroll
      for (int j = 0; j < 8; ++j) {
        float sc   = acc0[j] + acc1[j] + acc2[j];
        float dist = wq - 2.0f * sc;
        if (dist < minv[j]) { minv[j] = dist; mini[j] = kcol; }
      }
    }

    stage_wait();   // next tile resident; all waves done with current buffer
  }

  // Butterfly min-reduce across the 16 lanes of each half (mask<16 keeps
  // lanes within their half under xor). Tie -> lower index (matches argmin).
  #pragma unroll
  for (int j = 0; j < 8; ++j) {
    float v = minv[j];
    int   i = mini[j];
    for (int m = 1; m < 16; m <<= 1) {
      float ov = __shfl_xor(v, m, 32);
      int   oi = __shfl_xor(i, m, 32);
      if (ov < v || (ov == v && oi < i)) { v = ov; i = oi; }
    }
    minv[j] = v;
    mini[j] = i;
  }

  if (sub == 0) {
    int rbase = blockIdx.x * 128 + wave * 16 + half * 8;
    #pragma unroll
    for (int j = 0; j < 8; ++j) bestIdx[rbase + j] = mini[j];
  }
}

// ---- Kernel 5: gather selected codebook rows to output ---------------------
__global__ void k_gather(const float* __restrict__ w,
                         const int* __restrict__ idx,
                         float* __restrict__ out) {
  int n = blockIdx.x;
  int d = threadIdx.x;
  out[(size_t)n * DIMS + d] = w[(size_t)idx[n] * DIMS + d];
}

// ---------------------------------------------------------------------------
extern "C" void kernel_launch(void* const* d_in, const int* in_sizes, int n_in,
                              void* d_out, int out_size, void* d_ws, size_t ws_size,
                              hipStream_t stream) {
  (void)n_in; (void)out_size; (void)ws_size;
  const float* x     = (const float*)d_in[0];
  const float* w     = (const float*)d_in[1];
  const float* gamma = (const float*)d_in[2];
  const float* beta  = (const float*)d_in[3];

  int Dd = in_sizes[2];                 // 256
  if (Dd != DIMS) return;               // layout specialized for D=256
  int N = in_sizes[0] / Dd;             // 16384
  int K = in_sizes[1] / Dd;             // 8192

  const int RPB = 64;                   // rows per stats block
  int nPB = N / RPB;                    // 256

  // Workspace layout (all offsets keep 32B alignment for b128 vector loads)
  float* psum  = (float*)d_ws;                          // nPB * D
  float* psq   = psum + (size_t)nPB * Dd;               // nPB * D
  float* meanv = psq + (size_t)nPB * Dd;                // D
  float* rstdv = meanv + Dd;                            // D
  float* wsq   = rstdv + Dd;                            // K
  int*   bidx  = (int*)(wsq + K);                       // N
  unsigned short* wh = (unsigned short*)(bidx + N);     // K * D bf16-bits
  unsigned short* wl = wh + (size_t)K * Dd;             // K * D bf16-bits

  k_colstats_partial<<<nPB, 256, 0, stream>>>(x, psum, psq, RPB);
  k_colstats_final<<<1, 256, 0, stream>>>(psum, psq, nPB, 1.0f / (float)N,
                                          meanv, rstdv);
  k_split_w<<<K, 256, 0, stream>>>(w, wh, wl, wsq);

  // Dynamic LDS: 2 buffers x (hi + lo) x KT*DIMS bf16 = 128 KB (<= 320 KB WGP LDS)
  size_t ldsBytes = (size_t)2 * 2 * KT * DIMS * sizeof(unsigned short);
  k_argmin<<<N / 128, 256, ldsBytes, stream>>>(x, wh, wl, wsq, meanv, rstdv,
                                               gamma, beta, K, bidx);
  k_gather<<<N, 256, 0, stream>>>(w, bidx, (float*)d_out);
}
